// FP8Matmul_31344671326527
// MI455X (gfx1250) — 8-bit matrix operand order repaired and checked by reading, compiled, not run on hardware
//
#include <hip/hip_runtime.h>


#define E4M3_AMAX 216.0f

typedef __attribute__((ext_vector_type(16))) int   v16i;
typedef __attribute__((ext_vector_type(8)))  float v8f;

__device__ __forceinline__ unsigned char f32_to_e4m3_sw(float x) {
  union { float f; unsigned u; } v; v.f = x;
  const unsigned sgn = (v.u >> 24) & 0x80u;
  const unsigned au  = v.u & 0x7FFFFFFFu;
  if (au >= 0x3C800000u) {
    unsigned r = au + 0x0007FFFFu + ((au >> 20) & 1u);
    if (r > 0x43E00000u) r = 0x43E00000u;
    const unsigned e = (((r >> 23) - 127u + 7u) & 0xFu);
    const unsigned m = (r >> 20) & 0x7u;
    return (unsigned char)(sgn | (e << 3) | m);
  }
  float ax = fabsf(x);
  unsigned m = (unsigned)(ax * 512.0f + 0.5f);
  if (m > 7u) m = 7u;
  return (unsigned char)(sgn | m);
}

__device__ __forceinline__ unsigned pack4_e4m3(float a, float b, float c, float d,
                                               float s) {
  a = fminf(fmaxf(a * s, -E4M3_AMAX), E4M3_AMAX);
  b = fminf(fmaxf(b * s, -E4M3_AMAX), E4M3_AMAX);
  c = fminf(fmaxf(c * s, -E4M3_AMAX), E4M3_AMAX);
  d = fminf(fmaxf(d * s, -E4M3_AMAX), E4M3_AMAX);
#if __has_builtin(__builtin_amdgcn_cvt_pk_fp8_f32)
  int v = 0;
  v = __builtin_amdgcn_cvt_pk_fp8_f32(a, b, v, false);
  v = __builtin_amdgcn_cvt_pk_fp8_f32(c, d, v, true);
  return (unsigned)v;
#else
  return (unsigned)f32_to_e4m3_sw(a)
       | ((unsigned)f32_to_e4m3_sw(b) << 8)
       | ((unsigned)f32_to_e4m3_sw(c) << 16)
       | ((unsigned)f32_to_e4m3_sw(d) << 24);
#endif
}

__global__ __launch_bounds__(256) void fp8_quant_rm(const float* __restrict__ in,
                                                    unsigned* __restrict__ out,
                                                    const float* __restrict__ scale_p,
                                                    long long n4) {
  const float s = scale_p[0];
  long long i = (long long)blockIdx.x * blockDim.x + threadIdx.x;
  const long long stride = (long long)gridDim.x * blockDim.x;
  for (; i < n4; i += stride) {
    float4 f = ((const float4*)in)[i];
    out[i] = pack4_e4m3(f.x, f.y, f.z, f.w, s);
  }
}

__global__ __launch_bounds__(256) void fp8_quant_transpose(const float* __restrict__ in,
                                                           unsigned* __restrict__ out,
                                                           const float* __restrict__ scale_p,
                                                           int K, int N) {
  const float s = scale_p[0];
  const int k4cnt = K >> 2;
  const long long total = (long long)N * k4cnt;
  long long i = (long long)blockIdx.x * blockDim.x + threadIdx.x;
  const long long stride = (long long)gridDim.x * blockDim.x;
  for (; i < total; i += stride) {
    const int n  = (int)(i / k4cnt);
    const int k4 = (int)(i % k4cnt);
    const float* p = in + (long long)(k4 << 2) * N + n;
    const float a = p[0], b = p[(long long)N], c = p[2LL * N], d = p[3LL * N];
    out[i] = pack4_e4m3(a, b, c, d, s);
  }
}

#define BM 128
#define BN 128
#define BK 128
#define TILEB (128 * 128)

#if __has_builtin(__builtin_amdgcn_tensor_load_to_lds) && __has_builtin(__builtin_amdgcn_s_wait_tensorcnt)
#define USE_TDM 1
#else
#define USE_TDM 0
#endif

#if USE_TDM
typedef __attribute__((ext_vector_type(4))) unsigned int v4u;
typedef __attribute__((ext_vector_type(8))) int v8i_g;
typedef __attribute__((ext_vector_type(4))) int v4i_g;

__device__ __forceinline__ void tdm_load_tile_2d(unsigned lds_addr,
                                                 const unsigned char* gtile,
                                                 unsigned tensor_d0, unsigned tensor_d1,
                                                 unsigned long long d0_stride) {
  const unsigned long long ga = (unsigned long long)(const void*)gtile;
  v4u g0;
  g0[0] = 1u;
  g0[1] = lds_addr;
  g0[2] = (unsigned)ga;
  g0[3] = (unsigned)((ga >> 32) & 0x01FFFFFFull) | (2u << 30);
  v8i_g g1;
  g1[0] = 0;
  g1[1] = (int)((tensor_d0 & 0xFFFFu) << 16);
  g1[2] = (int)(((tensor_d0 >> 16) & 0xFFFFu) | ((tensor_d1 & 0xFFFFu) << 16));
  g1[3] = (int)(((tensor_d1 >> 16) & 0xFFFFu) | ((unsigned)BK << 16));
  g1[4] = (int)(unsigned)BM;
  g1[5] = (int)(unsigned)(d0_stride & 0xFFFFFFFFull);
  g1[6] = (int)(unsigned)((d0_stride >> 32) & 0xFFFFull);
  g1[7] = 0;
  v4i_g gz4 = {0, 0, 0, 0};
  v8i_g gz8 = {0, 0, 0, 0, 0, 0, 0, 0};
  __builtin_amdgcn_tensor_load_to_lds(g0, g1, gz4, gz4, gz8, 0);
}
#endif

__global__ __launch_bounds__(256) void fp8_wmma_gemm(const unsigned char* __restrict__ Aq,
                                                     const unsigned char* __restrict__ Bq,
                                                     const float* __restrict__ scale_p,
                                                     const float* __restrict__ scale1_p,
                                                     float* __restrict__ out,
                                                     int Rows, int Kdim, int Ncols) {
  extern __shared__ unsigned char lds[];
  unsigned char* sA = lds;
  unsigned char* sB = lds + 2 * TILEB;

  const int tid  = threadIdx.x;
  const int wave = tid >> 5;
  const int lane = tid & 31;
  const int half = lane >> 4;
  const int l15  = lane & 15;
  const int wm   = wave & 3;
  const int wn   = wave >> 2;

  const int rowBase = blockIdx.y * BM;
  const int colBase = blockIdx.x * BN;
  const long long K = (long long)Kdim;
  const int KT = Kdim / BK;

  const v8f vzero = {0.f, 0.f, 0.f, 0.f, 0.f, 0.f, 0.f, 0.f};
  v8f acc[2][4];
  #pragma unroll
  for (int i = 0; i < 2; ++i)
    #pragma unroll
    for (int j = 0; j < 4; ++j) acc[i][j] = vzero;

#if USE_TDM
  const unsigned ldsA0 = (unsigned)(unsigned long long)(const void*)sA;
  const unsigned ldsB0 = (unsigned)(unsigned long long)(const void*)sB;
  if (wave == 0) {
    tdm_load_tile_2d(ldsA0, Aq + (long long)rowBase * K, (unsigned)Kdim, (unsigned)Rows, (unsigned long long)Kdim);
    tdm_load_tile_2d(ldsB0, Bq + (long long)colBase * K, (unsigned)Kdim, (unsigned)Ncols, (unsigned long long)Kdim);
  }
#else
  {
    const int row = tid >> 1, off = (tid & 1) * 64;
    const uint4* ga = (const uint4*)(Aq + (long long)(rowBase + row) * K + off);
    const uint4* gb = (const uint4*)(Bq + (long long)(colBase + row) * K + off);
    uint4* da = (uint4*)(sA + row * BK + off);
    uint4* db = (uint4*)(sB + row * BK + off);
    #pragma unroll
    for (int i = 0; i < 4; ++i) { da[i] = ga[i]; db[i] = gb[i]; }
  }
#endif

  #pragma unroll 1
  for (int kt = 0; kt < KT; ++kt) {
    const int cur = kt & 1;
#if USE_TDM
    __builtin_amdgcn_s_wait_tensorcnt(0);
#endif
    __syncthreads();

    if (kt + 1 < KT) {
      const int nxt = cur ^ 1;
      const long long k0 = (long long)(kt + 1) * BK;
#if USE_TDM
      if (wave == 0) {
        tdm_load_tile_2d(ldsA0 + (unsigned)(nxt * TILEB), Aq + (long long)rowBase * K + k0,
                         (unsigned)Kdim, (unsigned)Rows, (unsigned long long)Kdim);
        tdm_load_tile_2d(ldsB0 + (unsigned)(nxt * TILEB), Bq + (long long)colBase * K + k0,
                         (unsigned)Kdim, (unsigned)Ncols, (unsigned long long)Kdim);
      }
#else
      const int row = tid >> 1, off = (tid & 1) * 64;
      const uint4* ga = (const uint4*)(Aq + (long long)(rowBase + row) * K + k0 + off);
      const uint4* gb = (const uint4*)(Bq + (long long)(colBase + row) * K + k0 + off);
      uint4* da = (uint4*)(sA + nxt * TILEB + row * BK + off);
      uint4* db = (uint4*)(sB + nxt * TILEB + row * BK + off);
      #pragma unroll
      for (int i = 0; i < 4; ++i) { da[i] = ga[i]; db[i] = gb[i]; }
#endif
    }

    const unsigned char* ap = sA + cur * TILEB + (wm * 32) * BK;
    const unsigned char* bp = sB + cur * TILEB + (wn * 64) * BK;

    v16i afr[2];
    #pragma unroll
    for (int i = 0; i < 2; ++i) {
      const unsigned char* p = ap + (i * 16 + l15) * BK + half * 16;
      v16i a;
      #pragma unroll
      for (int j = 0; j < 8; ++j) {
        uint2 d = *(const uint2*)(p + (j >> 1) * 32 + (j & 1) * 8);
        a[2 * j] = (int)d.x; a[2 * j + 1] = (int)d.y;
      }
      afr[i] = a;
    }
    v16i bfr[4];
    #pragma unroll
    for (int j = 0; j < 4; ++j) {
      const unsigned char* p = bp + (j * 16 + l15) * BK;
      v16i b;
      #pragma unroll
      for (int g = 0; g < 4; ++g) {
        uint4 d = *(const uint4*)(p + g * 32 + half * 16);
        b[4 * g + 0] = (int)d.x; b[4 * g + 1] = (int)d.y;
        b[4 * g + 2] = (int)d.z; b[4 * g + 3] = (int)d.w;
      }
      bfr[j] = b;
    }

    #pragma unroll
    for (int i = 0; i < 2; ++i)
      #pragma unroll
      for (int j = 0; j < 4; ++j)
        acc[i][j] = __builtin_amdgcn_wmma_f32_16x16x128_fp8_fp8(
            afr[i], bfr[j], (short)0, acc[i][j], false, false);

    __syncthreads();
  }

  const float inv = 1.0f / (scale_p[0] * scale1_p[0]);
  #pragma unroll
  for (int i = 0; i < 2; ++i)
    #pragma unroll
    for (int j = 0; j < 4; ++j)
      #pragma unroll
      for (int v = 0; v < 8; ++v) {
        const int r = rowBase + wm * 32 + i * 16 + half * 8 + v;
        const int c = colBase + wn * 64 + j * 16 + l15;
        out[(long long)r * Ncols + c] = acc[i][j][v] * inv;
      }
}

extern "C" void kernel_launch(void* const* d_in, const int* in_sizes, int n_in,
                              void* d_out, int out_size, void* d_ws, size_t ws_size,
                              hipStream_t stream) {
  (void)in_sizes; (void)n_in; (void)out_size; (void)ws_size;
  const float* input1 = (const float*)d_in[0];
  const float* input2 = (const float*)d_in[1];
  const float* scale  = (const float*)d_in[2];
  const float* scale1 = (const float*)d_in[3];
  float* out = (float*)d_out;

  const int Bsz = 4, M = 4096, Kd = 2048, Nd = 4096;
  const long long aElems = (long long)Bsz * M * Kd;
  unsigned char* Aq = (unsigned char*)d_ws;
  unsigned char* Bq = (unsigned char*)d_ws + aElems;

  {
    const long long n4 = aElems / 4;
    const int blocks = (int)((n4 + 255) / 256);
    fp8_quant_rm<<<blocks, 256, 0, stream>>>(input1, (unsigned*)Aq, scale, n4);
  }
  {
    const long long total = (long long)Nd * (Kd / 4);
    const int blocks = (int)((total + 255) / 256);
    fp8_quant_transpose<<<blocks, 256, 0, stream>>>(input2, (unsigned*)Bq, scale1, Kd, Nd);
  }
  {
    dim3 grid(Nd / BN, (Bsz * M) / BM);
    fp8_wmma_gemm<<<grid, 256, 4 * TILEB, stream>>>(Aq, Bq, scale, scale1, out,
                                                    Bsz * M, Kd, Nd);
  }
}
